// Cross_block_24867860644485
// MI455X (gfx1250) — compile-verified, matrix operand order checked by reading, not run on hardware
//
#include <hip/hip_runtime.h>

// ---------------------------------------------------------------------------
// Types for CDNA5 WMMA (wave32): v_wmma_f32_16x16x32_bf16
// ---------------------------------------------------------------------------
typedef __bf16 bf16_t;
typedef __attribute__((ext_vector_type(16))) __bf16 v16bf;
typedef __attribute__((ext_vector_type(8)))  __bf16 v8bf;
typedef __attribute__((ext_vector_type(4)))  __bf16 v4bf;
typedef __attribute__((ext_vector_type(8)))  float  v8f;
typedef __attribute__((ext_vector_type(4))) unsigned int u32x4;
typedef __attribute__((ext_vector_type(4))) int i32x4;
typedef __attribute__((ext_vector_type(8))) int i32x8;

#define B_SZ   8
#define SEQ    1024
#define DIM    128
#define HEADS  8
#define INNER  1024
#define HIDDEN 512
#define ROWS   (B_SZ * SEQ)   // 8192

__device__ __forceinline__ v8f wmma_bf16(v16bf a, v16bf b, v8f c) {
    // 8 args: (neg_a, A, neg_b, B, c_mod, C, reuse_a, reuse_b)
    return __builtin_amdgcn_wmma_f32_16x16x32_bf16(
        false, a, false, b, (short)0, c, false, false);
}

// Load one 16x32 bf16 fragment (A-layout; B-layout is the mirror with lane&15
// selecting the N column). Per ISA 7.12.2: lanes 0-15 hold K 0-7 & 16-23,
// lanes 16-31 hold K 8-15 & 24-31.
__device__ __forceinline__ v16bf load_frag(const bf16_t* __restrict__ base,
                                           int stride, int kb, int lane) {
    const bf16_t* p = base + (size_t)(lane & 15) * stride + kb + ((lane >> 4) << 3);
    union { v16bf v; v8bf h[2]; } u;
    u.h[0] = *(const v8bf*)(p);
    u.h[1] = *(const v8bf*)(p + 16);
    return u.v;
}

// LDS byte offset of a generic pointer to __shared__ (aperture: addr[31:0]).
__device__ __forceinline__ unsigned lds_off(const void* p) {
    return (unsigned)(unsigned long long)p;
}

// ---------------------------------------------------------------------------
// Tensor Data Mover: 2D tile (bf16) global -> LDS.  D# per ISA 08 §8.3-8.6:
//   group0: count=1 | lds_addr | global_addr[56:0] | type=2
//   group1: data_size=1(2B); tensor_dim0/1 huge (no OOB clipping);
//           tile_dim0/1; tensor_dim0_stride; groups 2/3 zero (<=2D tile).
// tile_d0 = contiguous elements per row, tile_d1 = rows, stride0 in elements.
// ---------------------------------------------------------------------------
__device__ __forceinline__ void tdm_load_2d(unsigned lds_addr, const void* gptr,
                                            unsigned tile_d0, unsigned tile_d1,
                                            unsigned long long stride0) {
    unsigned long long ga = (unsigned long long)gptr;
    u32x4 g0;
    g0[0] = 1u;                                   // count=1, user descriptor
    g0[1] = lds_addr;
    g0[2] = (unsigned)(ga & 0xFFFFFFFFull);
    g0[3] = (unsigned)((ga >> 32) & 0x1FFFFFFull) | (2u << 30);   // type=2

    const unsigned long long td0 = 1ull << 20;    // tensor dims: large
    const unsigned long long td1 = 1ull << 20;
    unsigned long long w0 = (1ull << 16)                          // data_size=2B
                          | ((td0 & 0xFFFFull) << 48);            // tensor_dim0[15:0]
    unsigned long long w1 = ((td0 >> 16) & 0xFFFFull)             // tensor_dim0[31:16]
                          | ((td1 & 0xFFFFFFFFull) << 16)         // tensor_dim1
                          | ((unsigned long long)(tile_d0 & 0xFFFFu) << 48);
    unsigned long long w2 = (unsigned long long)(tile_d1 & 0xFFFFu)
                          | ((stride0 & 0xFFFFFFFFull) << 32);    // stride0[31:0]
    unsigned long long w3 = (stride0 >> 32) & 0xFFFFull;          // stride0[47:32]
    union { i32x8 v; unsigned long long q[4]; } g1;
    g1.q[0] = w0; g1.q[1] = w1; g1.q[2] = w2; g1.q[3] = w3;
    i32x4 z4 = {0, 0, 0, 0};
#if defined(__clang_major__) && (__clang_major__ >= 23)
    i32x8 z8 = {0, 0, 0, 0, 0, 0, 0, 0};
    __builtin_amdgcn_tensor_load_to_lds(g0, g1.v, z4, z4, z8, 0);
#else
    __builtin_amdgcn_tensor_load_to_lds(g0, g1.v, z4, z4, 0);
#endif
}

// ---------------------------------------------------------------------------
// fp32 -> bf16 weight conversion
// ---------------------------------------------------------------------------
__global__ __launch_bounds__(256) void cvt_bf16_kernel(
    const float* __restrict__ in, bf16_t* __restrict__ out, int n) {
    int i = blockIdx.x * blockDim.x + threadIdx.x;
    if (i < n) out[i] = (bf16_t)in[i];
}

// ---------------------------------------------------------------------------
// LayerNorm over last dim (128) -> bf16.  One wave per row, wave32.
// ---------------------------------------------------------------------------
__global__ __launch_bounds__(256) void ln_bf16_kernel(
    const float* __restrict__ in, const float* __restrict__ w,
    const float* __restrict__ b, bf16_t* __restrict__ out, int rows) {
    int lane = threadIdx.x & 31;
    int row  = blockIdx.x * 8 + (threadIdx.x >> 5);
    if (row >= rows) return;
    const float4 v = ((const float4*)(in + (size_t)row * DIM))[lane];
    float s = v.x + v.y + v.z + v.w;
    s += __shfl_xor(s, 16, 32); s += __shfl_xor(s, 8, 32);
    s += __shfl_xor(s, 4, 32);  s += __shfl_xor(s, 2, 32);
    s += __shfl_xor(s, 1, 32);
    float mu = s * (1.0f / 128.0f);
    float dx = v.x - mu, dy = v.y - mu, dz = v.z - mu, dw = v.w - mu;
    float ss = dx * dx + dy * dy + dz * dz + dw * dw;
    ss += __shfl_xor(ss, 16, 32); ss += __shfl_xor(ss, 8, 32);
    ss += __shfl_xor(ss, 4, 32);  ss += __shfl_xor(ss, 2, 32);
    ss += __shfl_xor(ss, 1, 32);
    float rstd = rsqrtf(ss * (1.0f / 128.0f) + 1e-5f);
    const float4 wv = ((const float4*)w)[lane];
    const float4 bv = ((const float4*)b)[lane];
    v4bf o;
    o[0] = (bf16_t)(dx * rstd * wv.x + bv.x);
    o[1] = (bf16_t)(dy * rstd * wv.y + bv.y);
    o[2] = (bf16_t)(dz * rstd * wv.z + bv.z);
    o[3] = (bf16_t)(dw * rstd * wv.w + bv.w);
    ((v4bf*)(out + (size_t)row * DIM))[lane] = o;
}

// ---------------------------------------------------------------------------
// Generic WMMA GEMM:  C[m][n] = sum_k A[m][k] * W[n][k]   (A,W bf16 row-major)
// Block: 256 threads = 8 waves; wave computes 32x32 via 2x2 WMMA frags.
// ---------------------------------------------------------------------------
#define MODE_Q    0
#define MODE_KV   1
#define MODE_PROJ 2
#define MODE_FC1  3
#define MODE_FC2  4

template <int KDIM, int NCOLS, int MODE>
__global__ __launch_bounds__(256) void gemm_bf16_kernel(
    const bf16_t* __restrict__ A, const bf16_t* __restrict__ W,
    bf16_t* __restrict__ outb0, bf16_t* __restrict__ outb1,
    float*  __restrict__ outf,  const float* __restrict__ resid,
    const float* __restrict__ bias, const float* __restrict__ gamma) {
    int lane = threadIdx.x & 31;
    int wave = threadIdx.x >> 5;
    int m0 = blockIdx.x * 128 + (wave & 3) * 32;
    int n0 = blockIdx.y * 64  + (wave >> 2) * 32;

    v8f acc[2][2] = {};
    const bf16_t* a0p = A + (size_t)m0 * KDIM;
    const bf16_t* a1p = A + (size_t)(m0 + 16) * KDIM;
    const bf16_t* b0p = W + (size_t)n0 * KDIM;
    const bf16_t* b1p = W + (size_t)(n0 + 16) * KDIM;
    for (int kb = 0; kb < KDIM; kb += 32) {
        if (KDIM >= 512 && kb + 64 < KDIM) {   // warm next K-tile (global_prefetch_b8)
            __builtin_prefetch(a0p + (size_t)(lane & 15) * KDIM + kb + 64, 0, 1);
            __builtin_prefetch(b0p + (size_t)(lane & 15) * KDIM + kb + 64, 0, 1);
        }
        v16bf fa0 = load_frag(a0p, KDIM, kb, lane);
        v16bf fa1 = load_frag(a1p, KDIM, kb, lane);
        v16bf fb0 = load_frag(b0p, KDIM, kb, lane);
        v16bf fb1 = load_frag(b1p, KDIM, kb, lane);
        acc[0][0] = wmma_bf16(fa0, fb0, acc[0][0]);
        acc[0][1] = wmma_bf16(fa0, fb1, acc[0][1]);
        acc[1][0] = wmma_bf16(fa1, fb0, acc[1][0]);
        acc[1][1] = wmma_bf16(fa1, fb1, acc[1][1]);
    }

    int rbase = (lane >> 4) * 8;
    int cl    = lane & 15;
    for (int mi = 0; mi < 2; ++mi)
    for (int ni = 0; ni < 2; ++ni)
    for (int j = 0; j < 8; ++j) {
        int row = m0 + mi * 16 + rbase + j;
        int col = n0 + ni * 16 + cl;
        float val = acc[mi][ni][j];
        if (MODE == MODE_Q) {
            int bb = row >> 10, nn = row & 1023;
            int hh = col >> 7,  pp = col & 127;
            outb0[((((size_t)bb << 3) + hh) * SEQ + nn) * DIM + pp] = (bf16_t)val;
        } else if (MODE == MODE_KV) {
            int bb = row >> 10, mm = row & 1023;
            int s = col >> 10, hh = (col >> 7) & 7, pp = col & 127;
            size_t bh = ((size_t)bb << 3) + hh;
            if (s == 0) outb0[(bh * SEQ + mm) * DIM + pp] = (bf16_t)val;  // K: (B,H,M,P)
            else        outb1[(bh * DIM + pp) * SEQ + mm] = (bf16_t)val;  // V^T: (B,H,P,M)
        } else if (MODE == MODE_PROJ) {
            size_t idx = (size_t)row * DIM + col;
            outf[idx] = resid[idx] + (val + bias[col]) * gamma[col];
        } else if (MODE == MODE_FC1) {
            float r = val + bias[col];
            r = r > 0.0f ? r : 0.0f;
            outb0[(size_t)row * HIDDEN + col] = (bf16_t)r;
        } else { // MODE_FC2
            size_t idx = (size_t)row * DIM + col;
            outf[idx] = resid[idx] + (val + bias[col]) * gamma[col];
        }
    }
}

// ---------------------------------------------------------------------------
// Flash-style attention with TDM double-buffered K/V staging in LDS.
// Grid: (B*H) * (N/128) blocks; 8 waves/block, each wave owns 16 query rows.
// Per 32-key chunk: wave0 issues TENSOR_LOAD_TO_LDS for the next chunk
// (K: 32x128 contiguous; V^T: 128 rows x 32 keys, stride 1024), waits
// s_wait_tensorcnt on the current one; all waves then run 8 QK WMMAs,
// online softmax, P C-frag->A-frag transpose via LDS, 8 PV WMMAs.
// ---------------------------------------------------------------------------
__global__ __launch_bounds__(256) void attn_kernel(
    const bf16_t* __restrict__ q,   // (B,H,N,P)
    const bf16_t* __restrict__ kmat,// (B,H,M,P)
    const bf16_t* __restrict__ vT,  // (B,H,P,M)
    bf16_t* __restrict__ o)         // (B,N,H*P)
{
    __shared__ alignas(16) bf16_t kst[2][32][DIM];   // 2 x 8KB
    __shared__ alignas(16) bf16_t vst[2][DIM][32];   // 2 x 8KB
    __shared__ alignas(16) bf16_t plds[8][16 * 40];  // per-wave P tile
    int lane = threadIdx.x & 31;
    int wave = threadIdx.x >> 5;
    int bh   = blockIdx.x >> 3;
    int qt   = blockIdx.x & 7;
    int bb   = bh >> 3, hh = bh & 7;

    const bf16_t* qb = q    + (size_t)bh * SEQ * DIM;
    const bf16_t* kb = kmat + (size_t)bh * SEQ * DIM;
    const bf16_t* vb = vT   + (size_t)bh * DIM * SEQ;
    int q0 = qt * 128 + wave * 16;

    v16bf qf[4];
    for (int kk = 0; kk < 4; ++kk)
        qf[kk] = load_frag(qb + (size_t)q0 * DIM, DIM, kk * 32, lane);

    v8f oacc[8] = {};
    float mrow[8], lrow[8];
    for (int j = 0; j < 8; ++j) { mrow[j] = -1e30f; lrow[j] = 0.0f; }
    const float scale = 0.08838834764831845f;  // 1/sqrt(128)
    bf16_t* pl = plds[wave];
    int cl = lane & 15, rbase = (lane >> 4) * 8;

    // Prologue: stage chunk 0 into buffer 0.
    if (wave == 0) {
        tdm_load_2d(lds_off(&kst[0][0][0]), kb, 32 * DIM, 1, 32 * DIM); // contiguous 8KB
        tdm_load_2d(lds_off(&vst[0][0][0]), vb, 32, DIM, SEQ);
    }

    for (int mc = 0; mc < SEQ; mc += 32) {
        int buf = (mc >> 5) & 1;
        if (wave == 0) {
            if (mc + 32 < SEQ) {   // stage next chunk into the other buffer
                tdm_load_2d(lds_off(&kst[buf ^ 1][0][0]),
                            kb + (size_t)(mc + 32) * DIM, 32 * DIM, 1, 32 * DIM);
                tdm_load_2d(lds_off(&vst[buf ^ 1][0][0]),
                            vb + (mc + 32), 32, DIM, SEQ);
                __builtin_amdgcn_s_wait_tensorcnt(2);  // current chunk complete
            } else {
                __builtin_amdgcn_s_wait_tensorcnt(0);
            }
        }
        __syncthreads();   // staged K/V visible to all waves

        // ---- S = Q K^T for 32 keys (two 16x16 tiles) from LDS ----
        const bf16_t* kls = &kst[buf][0][0];
        v8f s0 = {}, s1 = {};
        for (int kk = 0; kk < 4; ++kk) {
            v16bf f0 = load_frag(kls,            DIM, kk * 32, lane);
            v16bf f1 = load_frag(kls + 16 * DIM, DIM, kk * 32, lane);
            s0 = wmma_bf16(qf[kk], f0, s0);
            s1 = wmma_bf16(qf[kk], f1, s1);
        }
        // ---- online softmax per query row ----
        float corr[8];
        for (int j = 0; j < 8; ++j) {
            float a = s0[j] * scale, c = s1[j] * scale;
            float r = fmaxf(a, c);
            r = fmaxf(r, __shfl_xor(r, 1, 32));
            r = fmaxf(r, __shfl_xor(r, 2, 32));
            r = fmaxf(r, __shfl_xor(r, 4, 32));
            r = fmaxf(r, __shfl_xor(r, 8, 32));
            float mnew = fmaxf(mrow[j], r);
            corr[j] = __expf(mrow[j] - mnew);
            mrow[j] = mnew;
            float p0 = __expf(a - mnew), p1 = __expf(c - mnew);
            s0[j] = p0; s1[j] = p1;
            float rs = p0 + p1;
            rs += __shfl_xor(rs, 1, 32);
            rs += __shfl_xor(rs, 2, 32);
            rs += __shfl_xor(rs, 4, 32);
            rs += __shfl_xor(rs, 8, 32);
            lrow[j] = lrow[j] * corr[j] + rs;
        }
        for (int nt = 0; nt < 8; ++nt)
            for (int j = 0; j < 8; ++j)
                oacc[nt][j] *= corr[j];
        // ---- transpose P (C-frag -> row-major 16x32 in LDS) ----
        for (int j = 0; j < 8; ++j) {
            pl[(rbase + j) * 40 + cl]      = (bf16_t)s0[j];
            pl[(rbase + j) * 40 + 16 + cl] = (bf16_t)s1[j];
        }
        v16bf pf;
        {
            const bf16_t* pp = pl + cl * 40 + ((lane >> 4) << 3);
            union { v16bf v; v8bf h[2]; } u;
            u.h[0] = *(const v8bf*)(pp);
            u.h[1] = *(const v8bf*)(pp + 16);
            pf = u.v;
        }
        // ---- O += P V  (V^T tile rows contiguous in keys) ----
        for (int nt = 0; nt < 8; ++nt) {
            const bf16_t* vp = &vst[buf][nt * 16 + cl][0] + ((lane >> 4) << 3);
            union { v16bf v; v8bf h[2]; } u;
            u.h[0] = *(const v8bf*)(vp);
            u.h[1] = *(const v8bf*)(vp + 16);
            oacc[nt] = wmma_bf16(pf, u.v, oacc[nt]);
        }
        __syncthreads();   // all reads of buf done before TDM overwrites it
    }
    // ---- finalize: divide by row sums, scatter to (B,N,H*P) ----
    float linv[8];
    for (int j = 0; j < 8; ++j) linv[j] = 1.0f / lrow[j];
    for (int nt = 0; nt < 8; ++nt)
        for (int j = 0; j < 8; ++j) {
            int row = q0 + rbase + j;
            int pp  = nt * 16 + cl;
            o[((size_t)bb * SEQ + row) * INNER + hh * DIM + pp] =
                (bf16_t)(oacc[nt][j] * linv[j]);
        }
}

// ---------------------------------------------------------------------------
// Host launcher
// ---------------------------------------------------------------------------
extern "C" void kernel_launch(void* const* d_in, const int* in_sizes, int n_in,
                              void* d_out, int out_size, void* d_ws, size_t ws_size,
                              hipStream_t stream) {
    (void)in_sizes; (void)n_in; (void)out_size; (void)ws_size;
    const float* x     = (const float*)d_in[0];
    const float* y     = (const float*)d_in[1];
    const float* nxw   = (const float*)d_in[2];
    const float* nxb   = (const float*)d_in[3];
    const float* nyw   = (const float*)d_in[4];
    const float* nyb   = (const float*)d_in[5];
    const float* qw    = (const float*)d_in[6];
    const float* kvw   = (const float*)d_in[7];
    const float* projw = (const float*)d_in[8];
    const float* projb = (const float*)d_in[9];
    const float* ls1   = (const float*)d_in[10];
    const float* now_  = (const float*)d_in[11];
    const float* nob   = (const float*)d_in[12];
    const float* fc1w  = (const float*)d_in[13];
    const float* fc1b  = (const float*)d_in[14];
    const float* fc2w  = (const float*)d_in[15];
    const float* fc2b  = (const float*)d_in[16];
    const float* ls2   = (const float*)d_in[17];

    char* p = (char*)d_ws;
    auto alloc = [&](size_t bytes) -> char* {
        char* r = p; p += (bytes + 255) & ~(size_t)255; return r;
    };
    bf16_t* xn   = (bf16_t*)alloc((size_t)ROWS * DIM * 2);
    bf16_t* yn   = (bf16_t*)alloc((size_t)ROWS * DIM * 2);
    bf16_t* qbuf = (bf16_t*)alloc((size_t)ROWS * INNER * 2);
    bf16_t* kbuf = (bf16_t*)alloc((size_t)ROWS * INNER * 2);
    bf16_t* vTb  = (bf16_t*)alloc((size_t)ROWS * INNER * 2);
    bf16_t* obuf = (bf16_t*)alloc((size_t)ROWS * INNER * 2);
    float*  x1   = (float*) alloc((size_t)ROWS * DIM * 4);
    bf16_t* x1n  = (bf16_t*)alloc((size_t)ROWS * DIM * 2);
    bf16_t* hbuf = (bf16_t*)alloc((size_t)ROWS * HIDDEN * 2);
    bf16_t* wq   = (bf16_t*)alloc((size_t)INNER * DIM * 2);
    bf16_t* wkv  = (bf16_t*)alloc((size_t)2 * INNER * DIM * 2);
    bf16_t* wpr  = (bf16_t*)alloc((size_t)DIM * INNER * 2);
    bf16_t* wf1  = (bf16_t*)alloc((size_t)HIDDEN * DIM * 2);
    bf16_t* wf2  = (bf16_t*)alloc((size_t)DIM * HIDDEN * 2);

    // 1) weights -> bf16
    cvt_bf16_kernel<<<(INNER * DIM + 255) / 256, 256, 0, stream>>>(qw, wq, INNER * DIM);
    cvt_bf16_kernel<<<(2 * INNER * DIM + 255) / 256, 256, 0, stream>>>(kvw, wkv, 2 * INNER * DIM);
    cvt_bf16_kernel<<<(DIM * INNER + 255) / 256, 256, 0, stream>>>(projw, wpr, DIM * INNER);
    cvt_bf16_kernel<<<(HIDDEN * DIM + 255) / 256, 256, 0, stream>>>(fc1w, wf1, HIDDEN * DIM);
    cvt_bf16_kernel<<<(DIM * HIDDEN + 255) / 256, 256, 0, stream>>>(fc2w, wf2, DIM * HIDDEN);

    // 2) LayerNorms
    ln_bf16_kernel<<<ROWS / 8, 256, 0, stream>>>(x, nxw, nxb, xn, ROWS);
    ln_bf16_kernel<<<ROWS / 8, 256, 0, stream>>>(y, nyw, nyb, yn, ROWS);

    // 3) Q and KV projections
    gemm_bf16_kernel<DIM, INNER, MODE_Q><<<dim3(ROWS / 128, INNER / 64), 256, 0, stream>>>(
        xn, wq, qbuf, nullptr, nullptr, nullptr, nullptr, nullptr);
    gemm_bf16_kernel<DIM, 2 * INNER, MODE_KV><<<dim3(ROWS / 128, 2 * INNER / 64), 256, 0, stream>>>(
        yn, wkv, kbuf, vTb, nullptr, nullptr, nullptr, nullptr);

    // 4) attention (TDM-staged K/V)
    attn_kernel<<<B_SZ * HEADS * (SEQ / 128), 256, 0, stream>>>(qbuf, kbuf, vTb, obuf);

    // 5) output projection + ls1 + residual -> x1 (f32)
    gemm_bf16_kernel<INNER, DIM, MODE_PROJ><<<dim3(ROWS / 128, DIM / 64), 256, 0, stream>>>(
        obuf, wpr, nullptr, nullptr, x1, x, projb, ls1);

    // 6) MLP
    ln_bf16_kernel<<<ROWS / 8, 256, 0, stream>>>(x1, now_, nob, x1n, ROWS);
    gemm_bf16_kernel<DIM, HIDDEN, MODE_FC1><<<dim3(ROWS / 128, HIDDEN / 64), 256, 0, stream>>>(
        x1n, wf1, hbuf, nullptr, nullptr, nullptr, fc1b, nullptr);
    gemm_bf16_kernel<HIDDEN, DIM, MODE_FC2><<<dim3(ROWS / 128, DIM / 64), 256, 0, stream>>>(
        hbuf, wf2, nullptr, nullptr, (float*)d_out, x1, fc2b, ls2);
}
